// WhileOpLstmLayer_61486751809786
// MI455X (gfx1250) — matrix operand order repaired and checked by reading, not compiled, not run on hardware
//
#include <hip/hip_runtime.h>

// LSTM: INPUT=1024, HIDDEN=1024, SEQ=256, BATCH=128
// bf16 WMMA (v_wmma_f32_16x16x32_bf16), f32 accumulate.
// B fragments staged into LDS by the Tensor Data Mover
// (tensor_load_to_lds, TENSORcnt, double-buffered 2x32KB), shared by the
// 8 waves of the block; WMMA stream reads B from LDS (ds_load) while TDM
// prefetches the next K-chunk.

#define HID   1024
#define NH4   4096
#define BATCH 128
#define SEQ   256

#define KCHUNK      256                      // K elements per TDM chunk
#define CHUNK_ELEMS (4 * 16 * KCHUNK)        // 16384 bf16 = 32 KB
#define NCHUNK      8                        // 4 for W-part + 4 for U-part

typedef __attribute__((ext_vector_type(16))) __bf16 v16bf;
typedef __attribute__((ext_vector_type(8)))  __bf16 v8bf;
typedef __attribute__((ext_vector_type(8)))  float  v8f;

typedef unsigned int u32;
typedef __attribute__((ext_vector_type(4))) u32 u32x4;
typedef __attribute__((ext_vector_type(8))) int i32x8;
typedef __attribute__((ext_vector_type(4))) int i32x4;

union AFrag { v16bf v; v8bf h[2]; };

__device__ __forceinline__ float sigmoid_f(float x) {
  return 1.0f / (1.0f + __expf(-x));
}

// ---------------------------------------------------------------------------
// TDM: load a (k=256) x (16 rows) x (4 gates) bf16 tile of a transposed
// weight matrix [4096 rows][1024 k] into LDS at byte offset lds_off.
//   dim0 = k (contiguous, tile 256 of tensor 1024)
//   dim1 = row within gate (tile 16, stride 1024 elems)
//   dim2 = gate            (tile 4,  stride 1024*1024 elems)
// D# bitfields per CDNA5 ISA 8.3-8.6.
// ---------------------------------------------------------------------------
__device__ __forceinline__ void tdm_load_slice(const __bf16* gsrc, u32 lds_off) {
  unsigned long long ga = (unsigned long long)(uintptr_t)gsrc;

  u32x4 g0;
  g0[0] = 1u;                                          // count=1 (valid, user)
  g0[1] = lds_off;                                     // lds_addr (bytes)
  g0[2] = (u32)ga;                                     // global_addr[31:0]
  g0[3] = (u32)((ga >> 32) & 0x01FFFFFFull)            // global_addr[56:32]
          | (2u << 30);                                // type = 2 ("image")

  i32x8 g1;
  g1[0] = 1 << 16;                 // data_size = 1 (2 bytes/elem)
  g1[1] = (int)(1024u << 16);      // tensor_dim0[15:0] = 1024 in bits[63:48]
  g1[2] = (int)(1024u << 16);      // dim0 hi=0; tensor_dim1[15:0]=1024 in [95:80]
  g1[3] = (int)(KCHUNK << 16);     // tile_dim0 = 256 in bits [127:112]
  g1[4] = (int)((4 << 16) | 16);   // tile_dim1 = 16, tile_dim2 = 4
  g1[5] = 1024;                    // tensor_dim0_stride[31:0] = 1024
  g1[6] = 0;                       // dim0_stride hi; dim1_stride[15:0]=0
  g1[7] = 0x10;                    // tensor_dim1_stride = 0x100000 (>>16)

  i32x4 g2;
  g2[0] = 4;                       // tensor_dim2 = 4 (gates)
  g2[1] = 0; g2[2] = 0; g2[3] = 0;
  i32x4 g3 = {0, 0, 0, 0};
  i32x8 g4 = {0, 0, 0, 0, 0, 0, 0, 0};   // unused extra group (clang-23 form)

  __builtin_amdgcn_tensor_load_to_lds(g0, g1, g2, g3, g4, 0);
}

// ---------------------------------------------------------------------------
// Prep: Wt[n*1024+k] = (bf16)W[k*4096+n]; same for U. Zero h0 and c.
// ---------------------------------------------------------------------------
__global__ __launch_bounds__(256) void lstm_prep(
    const float* __restrict__ w, const float* __restrict__ u,
    __bf16* __restrict__ w_t, __bf16* __restrict__ u_t,
    __bf16* __restrict__ h0, float* __restrict__ c)
{
  const int i = blockIdx.x * blockDim.x + threadIdx.x;
  if (i < NH4 * HID) {
    const int n = i >> 10;
    const int k = i & (HID - 1);
    const int src = k * NH4 + n;
    w_t[i] = (__bf16)w[src];
    u_t[i] = (__bf16)u[src];
  }
  if (i < BATCH * HID) {
    h0[i] = (__bf16)0.0f;
    c[i]  = 0.0f;
  }
}

// ---------------------------------------------------------------------------
// One recurrence step. Grid: 64 blocks (16 hidden cols each), 256 thr =
// 8 waves; wave w owns batch-tile m0 = 16*w. B fragments (4 gates x 16 cols)
// arrive via TDM into double-buffered LDS; all 8 waves read them as ds loads.
// ---------------------------------------------------------------------------
__global__ __launch_bounds__(256) void lstm_step(
    const float*  __restrict__ x_t,     // (128,1024) f32, this timestep
    const __bf16* __restrict__ w_t,     // (4096,1024) bf16, transposed
    const __bf16* __restrict__ u_t,     // (4096,1024) bf16, transposed
    const float*  __restrict__ bias,    // (4096) f32
    const __bf16* __restrict__ h_prev,  // (128,1024) bf16
    __bf16*       __restrict__ h_next,  // (128,1024) bf16
    float*        __restrict__ c,       // (128,1024) f32, in/out
    float*        __restrict__ h_out)   // (128,1024) f32 slice of d_out
{
  __shared__ __bf16 smem[2 * CHUNK_ELEMS];   // 64 KB double buffer

  const int wave = threadIdx.x >> 5;       // 0..7 -> batch tile
  const int lane = threadIdx.x & 31;
  const int half = lane >> 4;              // K-half selector (A/B layouts)
  const int l16  = lane & 15;
  const int m0   = wave << 4;              // batch-tile base row
  const int n0   = blockIdx.x << 4;        // hidden-tile base col

  v8f acc0 = {};  // input gate
  v8f acc1 = {};  // forget gate
  v8f acc2 = {};  // output gate
  v8f acc3 = {};  // cell candidate

  const int arow = m0 + l16;               // A-fragment row for this lane

  auto issue = [&](int cch) {
    const __bf16* base = (cch < 4) ? w_t : u_t;
    const int kb = (cch & 3) * KCHUNK;
    const __bf16* src = base + (size_t)n0 * HID + kb;
    const u32 lds_off =
        (u32)(uintptr_t)(const void*)&smem[(cch & 1) * CHUNK_ELEMS];
    tdm_load_slice(src, lds_off);
  };

  if (wave == 0) { issue(0); issue(1); }

  for (int cch = 0; cch < NCHUNK; ++cch) {
    if (wave == 0) {
      if (cch < NCHUNK - 1) __builtin_amdgcn_s_wait_tensorcnt(1);
      else                  __builtin_amdgcn_s_wait_tensorcnt(0);
    }
    __syncthreads();   // chunk cch resident in buf[cch&1]

    const __bf16* buf = &smem[(cch & 1) * CHUNK_ELEMS];
    const int kb0 = (cch & 3) * KCHUNK;

    if (cch < 4) {
      // ---- x_t @ W part (convert f32 -> bf16 on the fly) ----
      const float* xrow = x_t + (size_t)arow * HID + kb0 + half * 16;
      for (int kb = 0; kb < KCHUNK; kb += 32) {
        AFrag a;
        const float* p = xrow + kb;
        float4 f0 = *(const float4*)(p);
        float4 f1 = *(const float4*)(p + 4);
        float4 f2 = *(const float4*)(p + 8);
        float4 f3 = *(const float4*)(p + 12);
        a.v[0]  = (__bf16)f0.x; a.v[1]  = (__bf16)f0.y;
        a.v[2]  = (__bf16)f0.z; a.v[3]  = (__bf16)f0.w;
        a.v[4]  = (__bf16)f1.x; a.v[5]  = (__bf16)f1.y;
        a.v[6]  = (__bf16)f1.z; a.v[7]  = (__bf16)f1.w;
        a.v[8]  = (__bf16)f2.x; a.v[9]  = (__bf16)f2.y;
        a.v[10] = (__bf16)f2.z; a.v[11] = (__bf16)f2.w;
        a.v[12] = (__bf16)f3.x; a.v[13] = (__bf16)f3.y;
        a.v[14] = (__bf16)f3.z; a.v[15] = (__bf16)f3.w;

        const __bf16* bb = buf + kb + half * 16 + l16 * KCHUNK;
        v16bf b0 = *(const v16bf*)(bb + 0 * 16 * KCHUNK);
        v16bf b1 = *(const v16bf*)(bb + 1 * 16 * KCHUNK);
        v16bf b2 = *(const v16bf*)(bb + 2 * 16 * KCHUNK);
        v16bf b3 = *(const v16bf*)(bb + 3 * 16 * KCHUNK);

        acc0 = __builtin_amdgcn_wmma_f32_16x16x32_bf16(false, a.v, false, b0,
                                                       (short)0, acc0, false, false);
        acc1 = __builtin_amdgcn_wmma_f32_16x16x32_bf16(false, a.v, false, b1,
                                                       (short)0, acc1, false, false);
        acc2 = __builtin_amdgcn_wmma_f32_16x16x32_bf16(false, a.v, false, b2,
                                                       (short)0, acc2, false, false);
        acc3 = __builtin_amdgcn_wmma_f32_16x16x32_bf16(false, a.v, false, b3,
                                                       (short)0, acc3, false, false);
      }
    } else {
      // ---- h_prev @ U part (A already bf16) ----
      const __bf16* hrow = h_prev + (size_t)arow * HID + kb0 + half * 16;
      for (int kb = 0; kb < KCHUNK; kb += 32) {
        AFrag a;
        a.h[0] = *(const v8bf*)(hrow + kb);
        a.h[1] = *(const v8bf*)(hrow + kb + 8);

        const __bf16* bb = buf + kb + half * 16 + l16 * KCHUNK;
        v16bf b0 = *(const v16bf*)(bb + 0 * 16 * KCHUNK);
        v16bf b1 = *(const v16bf*)(bb + 1 * 16 * KCHUNK);
        v16bf b2 = *(const v16bf*)(bb + 2 * 16 * KCHUNK);
        v16bf b3 = *(const v16bf*)(bb + 3 * 16 * KCHUNK);

        acc0 = __builtin_amdgcn_wmma_f32_16x16x32_bf16(false, a.v, false, b0,
                                                       (short)0, acc0, false, false);
        acc1 = __builtin_amdgcn_wmma_f32_16x16x32_bf16(false, a.v, false, b1,
                                                       (short)0, acc1, false, false);
        acc2 = __builtin_amdgcn_wmma_f32_16x16x32_bf16(false, a.v, false, b2,
                                                       (short)0, acc2, false, false);
        acc3 = __builtin_amdgcn_wmma_f32_16x16x32_bf16(false, a.v, false, b3,
                                                       (short)0, acc3, false, false);
      }
    }

    __syncthreads();   // everyone done reading buf[cch&1]
    if (wave == 0 && cch + 2 < NCHUNK) issue(cch + 2);
  }

  // ---- Epilogue: gates, cell update, h writeback ----
  // C/D layout: VGPR v holds row m0 + v + 8*half, column n0 + l16.
  const int   col = n0 + l16;
  const float bi  = bias[col];
  const float bf_ = bias[col + HID];
  const float bo  = bias[col + 2 * HID];
  const float bc  = bias[col + 3 * HID];

#pragma unroll
  for (int v = 0; v < 8; ++v) {
    const int row = m0 + v + (half << 3);
    const int idx = row * HID + col;
    const float ig = sigmoid_f(acc0[v] + bi);
    const float fg = sigmoid_f(acc1[v] + bf_);
    const float og = sigmoid_f(acc2[v] + bo);
    const float cc = tanhf(acc3[v] + bc);
    const float cn = fg * c[idx] + ig * cc;
    c[idx] = cn;
    const float h = og * tanhf(cn);
    h_out[idx]  = h;
    h_next[idx] = (__bf16)h;
  }
}

// ---------------------------------------------------------------------------
extern "C" void kernel_launch(void* const* d_in, const int* in_sizes, int n_in,
                              void* d_out, int out_size, void* d_ws, size_t ws_size,
                              hipStream_t stream) {
  const float* x    = (const float*)d_in[0];  // (256,128,1024)
  const float* w    = (const float*)d_in[1];  // (1024,4096)
  const float* u    = (const float*)d_in[2];  // (1024,4096)
  const float* bias = (const float*)d_in[3];  // (4096)
  float* out = (float*)d_out;                 // (256,128,1024)

  char* ws = (char*)d_ws;
  size_t off = 0;
  __bf16* u_t = (__bf16*)(ws + off); off += (size_t)NH4 * HID * 2;      // 8 MB
  __bf16* w_t = (__bf16*)(ws + off); off += (size_t)NH4 * HID * 2;      // 8 MB
  __bf16* hb0 = (__bf16*)(ws + off); off += (size_t)BATCH * HID * 2;    // 256 KB
  __bf16* hb1 = (__bf16*)(ws + off); off += (size_t)BATCH * HID * 2;    // 256 KB
  float*  c   = (float*) (ws + off); off += (size_t)BATCH * HID * 4;    // 512 KB
  __bf16* hb[2] = { hb0, hb1 };

  lstm_prep<<<(NH4 * HID) / 256, 256, 0, stream>>>(w, u, w_t, u_t, hb0, c);

  for (int t = 0; t < SEQ; ++t) {
    lstm_step<<<dim3(HID / 16), 256, 0, stream>>>(
        x + (size_t)t * BATCH * HID, w_t, u_t, bias,
        hb[t & 1], hb[(t + 1) & 1], c,
        out + (size_t)t * BATCH * HID);
  }
}
